// RNN_2267742732464
// MI455X (gfx1250) — matrix operand order repaired and checked by reading, not compiled, not run on hardware
//
#include <hip/hip_runtime.h>
#include <hip/hip_bf16.h>

// ---- CDNA5 WMMA types -------------------------------------------------------
typedef __attribute__((ext_vector_type(16))) __bf16 v16bf;
typedef __attribute__((ext_vector_type(8)))  float  v8f;

union FragBF {           // 16 bf16 = 8 VGPRs, assembled from two 16B LDS loads
  uint4 q[2];
  v16bf v;
};

// packed f32x2 -> bf16x2, round-to-nearest (ties up): one v_add per input +
// one v_perm_b32 to merge the two high halves.
__device__ __forceinline__ unsigned pack2bf(float x, float y) {
#if __has_builtin(__builtin_amdgcn_cvt_pk_bf16_f32)
  return __builtin_bit_cast(unsigned, __builtin_amdgcn_cvt_pk_bf16_f32(x, y));
#elif __has_builtin(__builtin_amdgcn_perm)
  unsigned ux = __builtin_bit_cast(unsigned, x) + 0x8000u;
  unsigned uy = __builtin_bit_cast(unsigned, y) + 0x8000u;
  return __builtin_amdgcn_perm(uy, ux, 0x07060302u);  // [ux.hi16, uy.hi16]
#else
  unsigned ux = __builtin_bit_cast(unsigned, x) + 0x8000u;
  unsigned uy = __builtin_bit_cast(unsigned, y) + 0x8000u;
  return (ux >> 16) | (uy & 0xFFFF0000u);
#endif
}
// scalar f32 -> bf16: add + high-half store (folds to *_store_d16_hi_b16)
__device__ __forceinline__ unsigned short f2bf(float x) {
  return (unsigned short)((__builtin_bit_cast(unsigned, x) + 0x8000u) >> 16);
}

// hardware tanh (transcendental on gfx1250) with fallback
__device__ __forceinline__ float hw_tanh(float x) {
#if __has_builtin(__builtin_amdgcn_tanhf)
  return __builtin_amdgcn_tanhf(x);
#else
  return tanhf(x);
#endif
}

__device__ __forceinline__ void zero8(v8f& a) {
#pragma unroll
  for (int i = 0; i < 8; ++i) a[i] = 0.f;
}

// LDS byte offset of a generic pointer that points into __shared__
typedef __attribute__((address_space(3))) char lds_char;
template <typename T>
__device__ __forceinline__ unsigned ldsOff(T* p) {
  return (unsigned)(unsigned long long)(lds_char*)p;
}

// async global -> LDS copy of 16 bytes (per lane), tracked by ASYNCcnt
__device__ __forceinline__ void async_copy_b128(unsigned lds_dst, const void* gsrc) {
  asm volatile("global_load_async_to_lds_b128 %0, %1, off"
               :: "v"(lds_dst), "v"((unsigned long long)gsrc)
               : "memory");
}
__device__ __forceinline__ void async_wait0() {
  asm volatile("s_wait_asynccnt 0x0" ::: "memory");
}

// padded LDS row strides (elements): 144B/528B rows -> 36/132 dwords,
// gcd(.,64)==4 so 16 consecutive rows hit distinct banks; still 16B-aligned.
#define LDA 72
#define LDB 72
#define LWH 264
#define LH  264

// ---- W_emb: fp32 [k][n] -> bf16 [n][k], LDS-tiled 64x64 transpose -----------
// Both global phases fully coalesced; LDS pad 68 breaks transpose conflicts.
__global__ __launch_bounds__(256) void k_cvt_wemb(const float* __restrict__ W,
                                                  unsigned short* __restrict__ WT) {
  __shared__ unsigned short tile[64][68];
  const int tid = threadIdx.x;
  const int kb  = blockIdx.x;    // 0..127 (k tile of 64)
  const int nb  = blockIdx.y;    // 0..3   (n tile of 64)
  // read phase: rows k, consecutive n per thread group (coalesced)
#pragma unroll
  for (int it = 0; it < 16; ++it) {
    int idx = it * 256 + tid;            // 0..4095
    int kl = idx >> 6, nl = idx & 63;
    tile[kl][nl] = f2bf(W[(size_t)(kb * 64 + kl) * 256 + nb * 64 + nl]);
  }
  __syncthreads();
  // write phase: rows n, consecutive k per thread group (coalesced)
#pragma unroll
  for (int it = 0; it < 16; ++it) {
    int idx = it * 256 + tid;
    int nl = idx >> 6, kl = idx & 63;
    WT[(size_t)(nb * 64 + nl) * 8192 + kb * 64 + kl] = tile[kl][nl];
  }
}

__global__ void k_cvt_wenc(const float* __restrict__ W,
                           unsigned short* __restrict__ WxT,
                           unsigned short* __restrict__ WhT) {
  int idx = blockIdx.x * 256 + threadIdx.x;   // 256*256
  int n = idx & 255;
  int k = idx >> 8;
  WxT[n * 256 + k] = f2bf(W[(size_t)k * 256 + n]);
  WhT[n * 256 + k] = f2bf(W[(size_t)(k + 256) * 256 + n]);
}

// ---- GEMM1: xe[m][n] = x(16384x8192,f32) * W_emb(8192x256) + b_emb, bf16 out
// Software-pipelined: async-DMA B(kc+1) + register-prefetch A(kc+1) overlap the
// 8 WMMAs of chunk kc; one barrier per chunk, double-buffered LDS.
__global__ __launch_bounds__(256) void k_gemm_emb(
    const float* __restrict__ x, const unsigned short* __restrict__ WeT,
    const float* __restrict__ b_emb, unsigned short* __restrict__ xe) {
  __shared__ unsigned short ldsA[2][128 * LDA];   // 2 x 18 KB, bf16 [m][k]
  __shared__ unsigned short ldsB[2][64 * LDB];    // 2 x  9 KB, bf16 [n][k]
  const int tid  = threadIdx.x;
  const int lane = tid & 31;
  const int w    = tid >> 5;        // 8 waves, wave w owns M rows w*16..w*16+15
  const int nb   = blockIdx.x;      // 0..3  (N block of 64)
  const int mb   = blockIdx.y;      // 0..127 (M block of 128)

  v8f acc[4];
#pragma unroll
  for (int j = 0; j < 4; ++j) zero8(acc[j]);

  const float* xrow = x + (size_t)(mb * 128) * 8192;
  const unsigned short* Wrow = WeT + (size_t)(nb * 64) * 8192;

  // prologue: stage chunk 0 into buffer 0
#pragma unroll
  for (int it = 0; it < 2; ++it) {
    int idx = it * 256 + tid;
    int row = idx >> 3, q8 = idx & 7;
    async_copy_b128(ldsOff(&ldsB[0][row * LDB + q8 * 8]),
                    Wrow + (size_t)row * 8192 + q8 * 8);
  }
#pragma unroll
  for (int it = 0; it < 8; ++it) {
    int idx = it * 256 + tid;
    int row = idx >> 4, c4 = idx & 15;
    const float4 v = *(const float4*)(xrow + (size_t)row * 8192 + c4 * 4);
    unsigned* d = (unsigned*)&ldsA[0][row * LDA + c4 * 4];
    d[0] = pack2bf(v.x, v.y);
    d[1] = pack2bf(v.z, v.w);
  }
  async_wait0();
  __syncthreads();

  for (int kc = 0; kc < 128; ++kc) {          // K = 8192 in chunks of 64
    const int p = kc & 1, q = p ^ 1;
    const bool more = (kc + 1) < 128;
    float4 areg[8];
    if (more) {
      // async-DMA next B tile into the other buffer
#pragma unroll
      for (int it = 0; it < 2; ++it) {
        int idx = it * 256 + tid;
        int row = idx >> 3, q8 = idx & 7;
        async_copy_b128(ldsOff(&ldsB[q][row * LDB + q8 * 8]),
                        Wrow + (size_t)row * 8192 + (kc + 1) * 64 + q8 * 8);
      }
      // register-prefetch next A chunk (fp32)
#pragma unroll
      for (int it = 0; it < 8; ++it) {
        int idx = it * 256 + tid;
        int row = idx >> 4, c4 = idx & 15;
        areg[it] = *(const float4*)(xrow + (size_t)row * 8192 + (kc + 1) * 64 + c4 * 4);
      }
    }

    // compute chunk kc out of buffer p
#pragma unroll
    for (int ks = 0; ks < 2; ++ks) {          // two 16x16x32 K-steps per chunk
      FragBF a;
      int arow = w * 16 + (lane & 15);
      const char* ab = (const char*)&ldsA[p][0] + arow * (LDA * 2) +
                       (ks * 32 + ((lane >> 4) << 3)) * 2;
      a.q[0] = *(const uint4*)ab;
      a.q[1] = *(const uint4*)(ab + 32);
#pragma unroll
      for (int j = 0; j < 4; ++j) {
        FragBF b;
        int nrow = j * 16 + (lane & 15);
        const char* bb = (const char*)&ldsB[p][0] + nrow * (LDB * 2) +
                         (ks * 32 + ((lane >> 4) << 3)) * 2;
        b.q[0] = *(const uint4*)bb;
        b.q[1] = *(const uint4*)(bb + 32);
        acc[j] = __builtin_amdgcn_wmma_f32_16x16x32_bf16(
            false, a.v, false, b.v, (short)0, acc[j], false, false);
      }
    }

    if (more) {
      // convert + store prefetched A into the other buffer
#pragma unroll
      for (int it = 0; it < 8; ++it) {
        int idx = it * 256 + tid;
        int row = idx >> 4, c4 = idx & 15;
        unsigned* d = (unsigned*)&ldsA[q][row * LDA + c4 * 4];
        d[0] = pack2bf(areg[it].x, areg[it].y);
        d[1] = pack2bf(areg[it].z, areg[it].w);
      }
      async_wait0();
    }
    __syncthreads();
  }

  // epilogue: + b_emb, bf16 store (C layout: VGPR r -> M = r + 8*(lane>=16))
#pragma unroll
  for (int j = 0; j < 4; ++j) {
    int n = nb * 64 + j * 16 + (lane & 15);
    float bias = b_emb[n];
#pragma unroll
    for (int r = 0; r < 8; ++r) {
      int m = mb * 128 + w * 16 + r + ((lane >> 4) << 3);
      xe[(size_t)m * 256 + n] = f2bf(acc[j][r] + bias);
    }
  }
}

// ---- GEMM2: xp[m][n] = xe(bf16) * Wx + b_enc, fp32 out (into d_out) --------
__global__ __launch_bounds__(256) void k_gemm_enc(
    const unsigned short* __restrict__ xe, const unsigned short* __restrict__ WxT,
    const float* __restrict__ b_enc, float* __restrict__ xp) {
  __shared__ unsigned short ldsA[128 * LDA];
  __shared__ unsigned short ldsB[64 * LDB];
  const int tid  = threadIdx.x;
  const int lane = tid & 31;
  const int w    = tid >> 5;
  const int nb   = blockIdx.x;
  const int mb   = blockIdx.y;

  v8f acc[4];
#pragma unroll
  for (int j = 0; j < 4; ++j) zero8(acc[j]);

  for (int kc = 0; kc < 4; ++kc) {            // K = 256 in chunks of 64
#pragma unroll
    for (int it = 0; it < 4; ++it) {          // stage A via async DMA (bf16 copy)
      int idx = it * 256 + tid;
      int row = idx >> 3, q8 = idx & 7;
      async_copy_b128(ldsOff(&ldsA[row * LDA + q8 * 8]),
                      &xe[(size_t)(mb * 128 + row) * 256 + kc * 64 + q8 * 8]);
    }
#pragma unroll
    for (int it = 0; it < 2; ++it) {          // stage B via async DMA
      int idx = it * 256 + tid;
      int row = idx >> 3, q8 = idx & 7;
      async_copy_b128(ldsOff(&ldsB[row * LDB + q8 * 8]),
                      &WxT[(size_t)(nb * 64 + row) * 256 + kc * 64 + q8 * 8]);
    }
    async_wait0();
    __syncthreads();
#pragma unroll
    for (int ks = 0; ks < 2; ++ks) {
      FragBF a;
      int arow = w * 16 + (lane & 15);
      const char* ab = (const char*)ldsA + arow * (LDA * 2) +
                       (ks * 32 + ((lane >> 4) << 3)) * 2;
      a.q[0] = *(const uint4*)ab;
      a.q[1] = *(const uint4*)(ab + 32);
#pragma unroll
      for (int j = 0; j < 4; ++j) {
        FragBF b;
        int nrow = j * 16 + (lane & 15);
        const char* bb = (const char*)ldsB + nrow * (LDB * 2) +
                         (ks * 32 + ((lane >> 4) << 3)) * 2;
        b.q[0] = *(const uint4*)bb;
        b.q[1] = *(const uint4*)(bb + 32);
        acc[j] = __builtin_amdgcn_wmma_f32_16x16x32_bf16(
            false, a.v, false, b.v, (short)0, acc[j], false, false);
      }
    }
    __syncthreads();
  }
#pragma unroll
  for (int j = 0; j < 4; ++j) {
    int n = nb * 64 + j * 16 + (lane & 15);
    float bias = b_enc[n];
#pragma unroll
    for (int r = 0; r < 8; ++r) {
      int m = mb * 128 + w * 16 + r + ((lane >> 4) << 3);
      xp[(size_t)m * 256 + n] = acc[j][r] + bias;
    }
  }
}

// ---- Recurrence: h_t = tanh(xp_t + h_{t-1} * Wh), in-place over xp ---------
// Single 512-thread workgroup (16 waves). Uses CDNA5's 320KB LDS: Wh (135 KB,
// padded) AND h (33 KB) are LDS-resident, so every step is pure ds_load+WMMA.
__global__ __launch_bounds__(512) void k_rnn(
    const float* __restrict__ h0, const unsigned short* __restrict__ WhT,
    float* __restrict__ out) {
  __shared__ unsigned short ldsWh[256 * LWH];  // ~132 KB, bf16 [n][k] padded
  __shared__ unsigned short ldsH[64 * LH];     //  ~33 KB, bf16 [b][l] padded
  const int tid  = threadIdx.x;
  const int lane = tid & 31;
  const int w    = tid >> 5;       // 0..15
  const int mt   = w >> 2;         // M tile 0..3 (rows mt*16..)
  const int ntg  = (w & 3) * 4;    // 4 N tiles per wave

  // stage Wh into LDS via async DMA: 256 rows x 32 16B-chunks = 8192 / 512
#pragma unroll
  for (int it = 0; it < 16; ++it) {
    int idx = it * 512 + tid;
    int row = idx >> 5, c = idx & 31;
    async_copy_b128(ldsOff(&ldsWh[row * LWH + c * 8]),
                    &WhT[(size_t)row * 256 + c * 8]);
  }
  // load h0 (fp32) -> bf16 LDS
#pragma unroll
  for (int it = 0; it < 16; ++it) {
    int idx = it * 512 + tid;                 // pairs, 8192 total
    int m = idx >> 7, c = (idx & 127) * 2;
    float2 v = *(const float2*)(h0 + (size_t)m * 256 + c);
    *(unsigned*)&ldsH[m * LH + c] = pack2bf(v.x, v.y);
  }
  async_wait0();
  __syncthreads();

  for (int t = 0; t < 256; ++t) {
    v8f acc[4];
#pragma unroll
    for (int j = 0; j < 4; ++j) zero8(acc[j]);

#pragma unroll
    for (int kc = 0; kc < 8; ++kc) {          // K = 256 in 16x16x32 steps
      FragBF a;
      int arow = mt * 16 + (lane & 15);
      const char* ab = (const char*)ldsH + arow * (LH * 2) +
                       (kc * 32 + ((lane >> 4) << 3)) * 2;
      a.q[0] = *(const uint4*)ab;
      a.q[1] = *(const uint4*)(ab + 32);
#pragma unroll
      for (int j = 0; j < 4; ++j) {
        int n = (ntg + j) * 16 + (lane & 15);
        const char* bb = (const char*)ldsWh + n * (LWH * 2) +
                         (kc * 32 + ((lane >> 4) << 3)) * 2;
        FragBF b;
        b.q[0] = *(const uint4*)bb;
        b.q[1] = *(const uint4*)(bb + 32);
        acc[j] = __builtin_amdgcn_wmma_f32_16x16x32_bf16(
            false, a.v, false, b.v, (short)0, acc[j], false, false);
      }
    }
    __syncthreads();                          // all reads of ldsH done

#pragma unroll
    for (int j = 0; j < 4; ++j) {
      int n = (ntg + j) * 16 + (lane & 15);
#pragma unroll
      for (int r = 0; r < 8; ++r) {
        int m = mt * 16 + r + ((lane >> 4) << 3);
        size_t gi = ((size_t)t * 64 + m) * 256 + n;
        float v = hw_tanh(acc[j][r] + out[gi]); // out currently holds xp_t
        out[gi] = v;                            // overwrite with h_t
        ldsH[m * LH + n] = f2bf(v);             // state for next step
      }
    }
    __syncthreads();
  }
}

// ---- host side --------------------------------------------------------------
extern "C" void kernel_launch(void* const* d_in, const int* in_sizes, int n_in,
                              void* d_out, int out_size, void* d_ws, size_t ws_size,
                              hipStream_t stream) {
  (void)in_sizes; (void)n_in; (void)out_size; (void)ws_size;
  const float* x     = (const float*)d_in[0];   // (256,64,8192)
  const float* h0    = (const float*)d_in[1];   // (64,256)
  const float* W_emb = (const float*)d_in[2];   // (8192,256)
  const float* b_emb = (const float*)d_in[3];   // (256)
  const float* W_enc = (const float*)d_in[4];   // (512,256)
  const float* b_enc = (const float*)d_in[5];   // (256)
  float* out = (float*)d_out;                   // (256,64,256) fp32

  char* ws = (char*)d_ws;
  unsigned short* WeT = (unsigned short*)(ws);                          // 4 MiB
  unsigned short* WxT = (unsigned short*)(ws + (4u << 20));             // 128 KiB
  unsigned short* WhT = (unsigned short*)(ws + (4u << 20) + (128u << 10)); // 128 KiB
  unsigned short* xe  = (unsigned short*)(ws + (4u << 20) + (256u << 10)); // 8 MiB

  k_cvt_wemb<<<dim3(128, 4), 256, 0, stream>>>(W_emb, WeT);
  k_cvt_wenc<<<256, 256, 0, stream>>>(W_enc, WxT, WhT);
  k_gemm_emb<<<dim3(4, 128), 256, 0, stream>>>(x, WeT, b_emb, xe);
  k_gemm_enc<<<dim3(4, 128), 256, 0, stream>>>(xe, WxT, b_enc, out);
  k_rnn<<<1, 512, 0, stream>>>(h0, WhT, out);
}
